// MultiScaleMatcher_86208583565448
// MI455X (gfx1250) — matrix operand order repaired and checked by reading, not compiled, not run on hardware
//
#include <hip/hip_runtime.h>
#include <hip/hip_bf16.h>

typedef _Float16 f16;
typedef __attribute__((ext_vector_type(16))) _Float16 v16h;
typedef __attribute__((ext_vector_type(8)))  _Float16 v8h;
typedef __attribute__((ext_vector_type(8)))  float    v8f;

#define EPS_NORM 1e-12f
#define EPS_W    1e-6f
#define EPS_S    1e-8f
#define INV_REG  10.0f   // 1/REG, REG = 0.1

// ---------------- utility kernels ----------------

__global__ void zero_kernel(float* out, int n) {
    int i = blockIdx.x * blockDim.x + threadIdx.x;
    if (i < n) out[i] = 0.f;
}

// avg_pool2d (s,1) over (2,256,T,25)
__global__ void tpool_kernel(const float* __restrict__ in, float* __restrict__ out,
                             int Tin, int s, int total) {
    int Tout = Tin / s;
    for (int idx = blockIdx.x * blockDim.x + threadIdx.x; idx < total;
         idx += gridDim.x * blockDim.x) {
        int vv = idx % 25;
        int tt = (idx / 25) % Tout;
        int nc = idx / (25 * Tout);
        size_t base = ((size_t)nc * Tin + (size_t)tt * s) * 25 + vv;
        float sum = 0.f;
        for (int k = 0; k < s; ++k) sum += in[base + (size_t)k * 25];
        out[idx] = sum / (float)s;
    }
}

// mean of each of the 256 rows of X (blocks 0..255) and Y (blocks 256..511)
__global__ void rowmean_kernel(const float* __restrict__ X, const float* __restrict__ Y,
                               int M, float* __restrict__ xmean, float* __restrict__ ymean) {
    __shared__ float red[256];
    int rr = blockIdx.x;
    const float* src = (rr < 256) ? (X + (size_t)rr * M) : (Y + (size_t)(rr - 256) * M);
    float s = 0.f;
    for (int j = threadIdx.x; j < M; j += 256) s += src[j];
    red[threadIdx.x] = s; __syncthreads();
    for (int o = 128; o > 0; o >>= 1) {
        if (threadIdx.x < o) red[threadIdx.x] += red[threadIdx.x + o];
        __syncthreads();
    }
    if (threadIdx.x == 0) {
        float m = red[0] / (float)M;
        if (rr < 256) xmean[rr] = m; else ymean[rr - 256] = m;
    }
}

// per-column: L2 norm -> write normalized column transposed as f16 (M x 256 row-major),
// relu(dot with other-side row-mean)+eps -> raw weight; optionally init v=1
__global__ void colstats_kernel(const float* __restrict__ S, int M,
                                const float* __restrict__ omean,
                                f16* __restrict__ outT, float* __restrict__ raw,
                                float* __restrict__ vinit) {
    __shared__ float sm[256];
    sm[threadIdx.x] = omean[threadIdx.x];
    __syncthreads();
    int m = blockIdx.x * 256 + threadIdx.x;
    if (m >= M) return;
    float sumsq = 0.f, dot = 0.f;
    for (int cc = 0; cc < 256; ++cc) {
        float val = S[(size_t)cc * M + m];
        sumsq += val * val;
        dot   += val * sm[cc];
    }
    float inv = 1.f / fmaxf(sqrtf(sumsq), EPS_NORM);
    f16* orow = outT + (size_t)m * 256;
    for (int cc = 0; cc < 256; ++cc)
        orow[cc] = (f16)(S[(size_t)cc * M + m] * inv);
    raw[m] = fmaxf(dot, 0.f) + EPS_W;
    if (vinit) vinit[m] = 1.0f;
}

__global__ void sums_kernel(const float* __restrict__ rraw, const float* __restrict__ craw,
                            int M, float* __restrict__ sums) {
    __shared__ float red[256];
    float sr = 0.f, sc = 0.f;
    for (int i = threadIdx.x; i < M; i += 256) { sr += rraw[i]; sc += craw[i]; }
    red[threadIdx.x] = sr; __syncthreads();
    for (int o = 128; o > 0; o >>= 1) {
        if (threadIdx.x < o) red[threadIdx.x] += red[threadIdx.x + o];
        __syncthreads();
    }
    if (threadIdx.x == 0) sums[0] = red[0];
    __syncthreads();
    red[threadIdx.x] = sc; __syncthreads();
    for (int o = 128; o > 0; o >>= 1) {
        if (threadIdx.x < o) red[threadIdx.x] += red[threadIdx.x + o];
        __syncthreads();
    }
    if (threadIdx.x == 0) sums[1] = red[0];
}

// ---------------- WMMA GEMM: Kmat = exp((A.Bt - 1) * 10), fp16 out ----------------
// A: (M x 256) f16 row-major = Xn^T ; B: (M x 256) f16 row-major = Yn^T
// 8 waves per block; block covers 64 (M) x 32 (N); one 16x16 tile per wave.
__global__ void gemm_exp_kernel(const f16* __restrict__ A, const f16* __restrict__ B,
                                f16* __restrict__ Kmat, int M) {
    int w    = threadIdx.x >> 5;
    int lane = threadIdx.x & 31;
    int hl   = lane >> 4;     // half-wave select
    int l    = lane & 15;
    int tileM = blockIdx.x * 64 + (w >> 1) * 16;
    int tileN = blockIdx.y * 32 + (w & 1) * 16;
    const f16* Arow = A + (size_t)(tileM + l) * 256;
    const f16* Brow = B + (size_t)(tileN + l) * 256;
    v8f acc = {};
#pragma unroll
    for (int k0 = 0; k0 < 256; k0 += 32) {
        // A fragment (16-bit A 16x32): lanes<16 hold K[0..7],K[16..23]; lanes>=16 K[8..15],K[24..31]
        v8h a0 = *(const v8h*)(Arow + k0 + hl * 8);
        v8h a1 = *(const v8h*)(Arow + k0 + 16 + hl * 8);
        // B fragment (16-bit B 32x16): lanes<16 hold K[0..7],K[16..23] of col l; lanes>=16 K[8..15],K[24..31]
        v8h b0 = *(const v8h*)(Brow + k0 + hl * 8);
        v8h b1 = *(const v8h*)(Brow + k0 + 16 + hl * 8);
        v16h av = __builtin_shufflevector(a0, a1, 0,1,2,3,4,5,6,7,8,9,10,11,12,13,14,15);
        v16h bv = __builtin_shufflevector(b0, b1, 0,1,2,3,4,5,6,7,8,9,10,11,12,13,14,15);
        acc = __builtin_amdgcn_wmma_f32_16x16x32_f16(false, av, false, bv,
                                                     (short)0, acc, false, false);
    }
    // C/D layout: VGPR r -> row = r + 8*hl, col = lane&15
    int j = tileN + l;
#pragma unroll
    for (int r = 0; r < 8; ++r) {
        int i = tileM + hl * 8 + r;
        Kmat[(size_t)i * M + j] = (f16)__expf(INV_REG * (acc[r] - 1.f));
    }
}

// ---------------- Sinkhorn ----------------

// u = r / (K @ v + eps) : one wave per row, vectorized fp16 row streaming
__global__ void u_update_kernel(const f16* __restrict__ Kmat, const float* __restrict__ v,
                                const float* __restrict__ rraw, const float* __restrict__ sums,
                                float* __restrict__ u, int M) {
    int row  = blockIdx.x * 8 + (threadIdx.x >> 5);
    int lane = threadIdx.x & 31;
    const f16* Krow = Kmat + (size_t)row * M;
    float acc = 0.f;
    for (int j0 = lane * 8; j0 < M; j0 += 256) {
        v8h kv = *(const v8h*)(Krow + j0);
#pragma unroll
        for (int r = 0; r < 8; ++r) acc += (float)kv[r] * v[j0 + r];
    }
#pragma unroll
    for (int o = 16; o > 0; o >>= 1) acc += __shfl_xor(acc, o, 32);
    if (lane == 0) u[row] = (rraw[row] / sums[0]) / (acc + EPS_S);
}

// v = c / (K^T @ u + eps) : one thread per column, coalesced column streaming
__global__ void v_update_kernel(const f16* __restrict__ Kmat, const float* __restrict__ u,
                                const float* __restrict__ craw, const float* __restrict__ sums,
                                float* __restrict__ v, int M) {
    int j = blockIdx.x * 256 + threadIdx.x;
    if (j >= M) return;
    float acc = 0.f;
    for (int i = 0; i < M; ++i) acc += (float)Kmat[(size_t)i * M + j] * u[i];
    v[j] = (craw[j] / sums[1]) / (acc + EPS_S);
}

// score += weight * sum_ij (1 - D) * u_i K_ij v_j ; 1-D recovered as 1 + 0.1*ln(K)
__global__ void score_kernel(const f16* __restrict__ Kmat, const float* __restrict__ u,
                             const float* __restrict__ v, int M, float weight,
                             float* __restrict__ out) {
    __shared__ float red[256];
    int j = blockIdx.x * 256 + threadIdx.x;
    float acc = 0.f;
    if (j < M) {
        for (int i = 0; i < M; ++i) {
            float k = (float)Kmat[(size_t)i * M + j];
            if (k > 0.f) acc += (1.f + 0.1f * __logf(k)) * k * u[i];
        }
        acc *= v[j];
    }
    red[threadIdx.x] = acc; __syncthreads();
    for (int o = 128; o > 0; o >>= 1) {
        if (threadIdx.x < o) red[threadIdx.x] += red[threadIdx.x + o];
        __syncthreads();
    }
    if (threadIdx.x == 0) atomicAdd(out, weight * red[0]);
}

// ---------------- cross-scale (gpool) path ----------------

// feats[a][c*2+n] = mean over (T,V) of array a at (n,c)
__global__ void gpool_kernel(const float* __restrict__ x1, const float* __restrict__ x2,
                             const float* __restrict__ x3, const float* __restrict__ y1,
                             const float* __restrict__ y2, const float* __restrict__ y3,
                             float* __restrict__ feats) {
    const float* arrs[6] = { x1, x2, x3, y1, y2, y3 };
    int a  = blockIdx.x;
    int nc = blockIdx.y;                    // n*256 + c
    const float* src = arrs[a] + (size_t)nc * 3200;
    __shared__ float red[256];
    float s = 0.f;
    for (int t = threadIdx.x; t < 3200; t += 256) s += src[t];
    red[threadIdx.x] = s; __syncthreads();
    for (int o = 128; o > 0; o >>= 1) {
        if (threadIdx.x < o) red[threadIdx.x] += red[threadIdx.x + o];
        __syncthreads();
    }
    if (threadIdx.x == 0) {
        int n = nc >> 8, c = nc & 255;
        feats[a * 512 + c * 2 + n] = red[0] / 3200.f;
    }
}

// whole EMD on 256x2 features; one wave per (i,j) pair, i != j
__global__ void tiny_emd_kernel(const float* __restrict__ feats, float* __restrict__ out) {
    const int pi[6] = { 0, 0, 1, 1, 2, 2 };
    const int pj[6] = { 1, 2, 0, 2, 0, 1 };
    const float* X = feats + pi[blockIdx.x] * 512;
    const float* Y = feats + (3 + pj[blockIdx.x]) * 512;
    int lane = threadIdx.x;
    float s00 = 0, s01 = 0, s10 = 0, s11 = 0, qx0 = 0, qx1 = 0, qy0 = 0, qy1 = 0;
    for (int c = lane; c < 256; c += 32) {
        float x0 = X[c * 2], x1 = X[c * 2 + 1];
        float y0 = Y[c * 2], y1 = Y[c * 2 + 1];
        s00 += x0 * y0; s01 += x0 * y1; s10 += x1 * y0; s11 += x1 * y1;
        qx0 += x0 * x0; qx1 += x1 * x1; qy0 += y0 * y0; qy1 += y1 * y1;
    }
#pragma unroll
    for (int o = 16; o > 0; o >>= 1) {
        s00 += __shfl_xor(s00, o, 32); s01 += __shfl_xor(s01, o, 32);
        s10 += __shfl_xor(s10, o, 32); s11 += __shfl_xor(s11, o, 32);
        qx0 += __shfl_xor(qx0, o, 32); qx1 += __shfl_xor(qx1, o, 32);
        qy0 += __shfl_xor(qy0, o, 32); qy1 += __shfl_xor(qy1, o, 32);
    }
    if (lane == 0) {
        float nx0 = fmaxf(sqrtf(qx0), EPS_NORM), nx1 = fmaxf(sqrtf(qx1), EPS_NORM);
        float ny0 = fmaxf(sqrtf(qy0), EPS_NORM), ny1 = fmaxf(sqrtf(qy1), EPS_NORM);
        float S00 = s00 / (nx0 * ny0), S01 = s01 / (nx0 * ny1);
        float S10 = s10 / (nx1 * ny0), S11 = s11 / (nx1 * ny1);
        // weights: X^T @ mean(Y, axis=1) etc.
        float rr0 = fmaxf(0.5f * (s00 + s01), 0.f) + EPS_W;
        float rr1 = fmaxf(0.5f * (s10 + s11), 0.f) + EPS_W;
        float cc0 = fmaxf(0.5f * (s00 + s10), 0.f) + EPS_W;
        float cc1 = fmaxf(0.5f * (s01 + s11), 0.f) + EPS_W;
        float r0 = rr0 / (rr0 + rr1), r1 = rr1 / (rr0 + rr1);
        float c0 = cc0 / (cc0 + cc1), c1 = cc1 / (cc0 + cc1);
        float K00 = __expf(INV_REG * (S00 - 1.f)), K01 = __expf(INV_REG * (S01 - 1.f));
        float K10 = __expf(INV_REG * (S10 - 1.f)), K11 = __expf(INV_REG * (S11 - 1.f));
        float u0 = 1.f, u1 = 1.f, v0 = 1.f, v1 = 1.f;
        for (int it = 0; it < 100; ++it) {
            u0 = r0 / (K00 * v0 + K01 * v1 + EPS_S);
            u1 = r1 / (K10 * v0 + K11 * v1 + EPS_S);
            v0 = c0 / (K00 * u0 + K10 * u1 + EPS_S);
            v1 = c1 / (K01 * u0 + K11 * u1 + EPS_S);
        }
        float score = S00 * u0 * K00 * v0 + S01 * u0 * K01 * v1 +
                      S10 * u1 * K10 * v0 + S11 * u1 * K11 * v1;
        atomicAdd(out, score);
    }
}

// ---------------- host orchestration ----------------

extern "C" void kernel_launch(void* const* d_in, const int* in_sizes, int n_in,
                              void* d_out, int out_size, void* d_ws, size_t ws_size,
                              hipStream_t stream) {
    const float* x1 = (const float*)d_in[0];
    const float* x2 = (const float*)d_in[1];
    const float* x3 = (const float*)d_in[2];
    const float* y1 = (const float*)d_in[3];
    const float* y2 = (const float*)d_in[4];
    const float* y3 = (const float*)d_in[5];
    float* out = (float*)d_out;
    (void)in_sizes; (void)n_in; (void)ws_size;

    size_t off = 0;
    char* base = (char*)d_ws;
    auto wsp = [&](size_t bytes) -> char* {
        char* p = base + off;
        off += (bytes + 255) & ~(size_t)255;
        return p;
    };
    f16*   Kmat  = (f16*)wsp(6400ull * 6400ull * sizeof(f16));   // 81.92 MB (fits L2)
    f16*   Afp   = (f16*)wsp(6400ull * 256 * sizeof(f16));
    f16*   Bfp   = (f16*)wsp(6400ull * 256 * sizeof(f16));
    float* px2   = (float*)wsp(819200ull * 4);
    float* py2   = (float*)wsp(819200ull * 4);
    float* px4   = (float*)wsp(409600ull * 4);
    float* py4   = (float*)wsp(409600ull * 4);
    float* xmean = (float*)wsp(256 * 4);
    float* ymean = (float*)wsp(256 * 4);
    float* rraw  = (float*)wsp(6400 * 4);
    float* craw  = (float*)wsp(6400 * 4);
    float* uvec  = (float*)wsp(6400 * 4);
    float* vvec  = (float*)wsp(6400 * 4);
    float* sums  = (float*)wsp(2 * 4);
    float* feats = (float*)wsp(6 * 512 * 4);

    zero_kernel<<<1, 64, 0, stream>>>(out, out_size);

    // temporal pooling of x1/y1 for s=2,4
    tpool_kernel<<<1024, 256, 0, stream>>>(x1, px2, 128, 2, 819200);
    tpool_kernel<<<1024, 256, 0, stream>>>(y1, py2, 128, 2, 819200);
    tpool_kernel<<<1024, 256, 0, stream>>>(x1, px4, 128, 4, 409600);
    tpool_kernel<<<1024, 256, 0, stream>>>(y1, py4, 128, 4, 409600);

    struct Pair { const float* X; const float* Y; int M; float w; };
    // emd(x1,y1) appears twice in the reference (scale loop + temporal s=1) -> weight 2
    Pair pairs[5] = {
        { x1,  y1,  6400, 2.f },
        { x2,  y2,  6400, 1.f },
        { x3,  y3,  6400, 1.f },
        { px2, py2, 3200, 1.f },
        { px4, py4, 1600, 1.f },
    };

    for (int p = 0; p < 5; ++p) {
        const float* X = pairs[p].X;
        const float* Y = pairs[p].Y;
        int M = pairs[p].M;
        int gb = (M + 255) / 256;

        rowmean_kernel<<<512, 256, 0, stream>>>(X, Y, M, xmean, ymean);
        colstats_kernel<<<gb, 256, 0, stream>>>(X, M, ymean, Afp, rraw, nullptr);
        colstats_kernel<<<gb, 256, 0, stream>>>(Y, M, xmean, Bfp, craw, vvec);
        sums_kernel<<<1, 256, 0, stream>>>(rraw, craw, M, sums);

        gemm_exp_kernel<<<dim3(M / 64, M / 32), 256, 0, stream>>>(Afp, Bfp, Kmat, M);

        for (int it = 0; it < 100; ++it) {
            u_update_kernel<<<M / 8, 256, 0, stream>>>(Kmat, vvec, rraw, sums, uvec, M);
            v_update_kernel<<<gb, 256, 0, stream>>>(Kmat, uvec, craw, sums, vvec, M);
        }
        score_kernel<<<gb, 256, 0, stream>>>(Kmat, uvec, vvec, M, pairs[p].w, out);
    }

    // cross-scale 2x2 EMDs on gpooled features
    gpool_kernel<<<dim3(6, 512), 256, 0, stream>>>(x1, x2, x3, y1, y2, y3, feats);
    tiny_emd_kernel<<<6, 32, 0, stream>>>(feats, out);
}
